// MultiPhasePhasor_37082747633922
// MI455X (gfx1250) — matrix operand order repaired and checked by reading, not compiled, not run on hardware
//
#include <hip/hip_runtime.h>
#include <hip/hip_bf16.h>

// ---------------------------------------------------------------------------
// Problem constants (from reference)
// ---------------------------------------------------------------------------
#define DIMD   2048
#define NPH    8
#define BB     4
#define LL     4096
#define MTOT   (BB * LL)      // 16384 tokens
#define CHUNK  64
#define NCHUNK (LL / CHUNK)   // 64

typedef __bf16        v16bf __attribute__((ext_vector_type(16)));
typedef float         v8f   __attribute__((ext_vector_type(8)));
typedef unsigned int  v4u   __attribute__((ext_vector_type(4)));
typedef int           v4i   __attribute__((vector_size(16)));   // builtin param type

// CDNA5 async global->LDS path (guarded: falls back to reg-staged copies)
#if __has_builtin(__builtin_amdgcn_global_load_async_to_lds_b128) && \
    __has_builtin(__builtin_amdgcn_s_wait_asynccnt)
#define USE_ASYNC_LDS 1
#else
#define USE_ASYNC_LDS 0
#endif

#if USE_ASYNC_LDS
static __device__ inline void async_copy16(const __bf16* g, __bf16* l) {
    __builtin_amdgcn_global_load_async_to_lds_b128(
        (v4i*)g, (v4i*)l, /*offset=*/0, /*cpol=*/0);
}
#endif

// ---------------------------------------------------------------------------
// K0: fp32 -> bf16 conversion (4 floats -> 8 bytes per thread)
// ---------------------------------------------------------------------------
__global__ __launch_bounds__(256) void cvt_bf16_kernel(const float* __restrict__ in,
                                                       __bf16* __restrict__ out,
                                                       long n4) {
    long i = (long)blockIdx.x * blockDim.x + threadIdx.x;
    if (i >= n4) return;
    float4 f = ((const float4*)in)[i];
    union { __bf16 h[4]; unsigned long long u; } u;
    u.h[0] = (__bf16)f.x; u.h[1] = (__bf16)f.y;
    u.h[2] = (__bf16)f.z; u.h[3] = (__bf16)f.w;
    ((unsigned long long*)out)[i] = u.u;
}

// ---------------------------------------------------------------------------
// K1: per-token phasor scalar  s[m] = sum_p (cos(ph)+sin(ph)) * amp
//     one wave32 per token, 8 tokens per 256-thread block
// ---------------------------------------------------------------------------
__global__ __launch_bounds__(256) void phase_amp_kernel(const float* __restrict__ x,
                                                        const float* __restrict__ Wp,
                                                        const float* __restrict__ bp,
                                                        const float* __restrict__ Wa,
                                                        const float* __restrict__ ba,
                                                        float* __restrict__ s) {
    const int wave = threadIdx.x >> 5;
    const int lane = threadIdx.x & 31;
    const int m = blockIdx.x * 8 + wave;
    if (m >= MTOT) return;

    const float* xr = x + (size_t)m * DIMD;
    float ap[NPH], aa[NPH];
#pragma unroll
    for (int p = 0; p < NPH; ++p) { ap[p] = 0.f; aa[p] = 0.f; }

    for (int kk = 0; kk < DIMD / 32; ++kk) {
        const int k = kk * 32 + lane;          // coalesced across the wave
        const float xv = xr[k];
        const float4* wp4 = (const float4*)(Wp + (size_t)k * NPH);
        const float4* wa4 = (const float4*)(Wa + (size_t)k * NPH);
        float4 p0 = wp4[0], p1 = wp4[1];
        float4 a0 = wa4[0], a1 = wa4[1];
        ap[0] += xv * p0.x; ap[1] += xv * p0.y; ap[2] += xv * p0.z; ap[3] += xv * p0.w;
        ap[4] += xv * p1.x; ap[5] += xv * p1.y; ap[6] += xv * p1.z; ap[7] += xv * p1.w;
        aa[0] += xv * a0.x; aa[1] += xv * a0.y; aa[2] += xv * a0.z; aa[3] += xv * a0.w;
        aa[4] += xv * a1.x; aa[5] += xv * a1.y; aa[6] += xv * a1.z; aa[7] += xv * a1.w;
    }
#pragma unroll
    for (int p = 0; p < NPH; ++p) {
        for (int off = 16; off; off >>= 1) {
            ap[p] += __shfl_xor(ap[p], off);
            aa[p] += __shfl_xor(aa[p], off);
        }
    }
    if (lane == 0) {
        const float PI = 3.14159265358979323846f;
        float acc = 0.f;
#pragma unroll
        for (int p = 0; p < NPH; ++p) {
            float ph = tanhf(ap[p] + bp[p]) * PI;
            float a  = aa[p] + ba[p];
            float sp = (a > 20.f) ? a : log1pf(expf(a));
            acc += (cosf(ph) + sinf(ph)) * (sp + 0.1f);
        }
        s[m] = acc;
    }
}

// ---------------------------------------------------------------------------
// K2: bf16 WMMA GEMM, block tile 128x128, BK=64, double-buffered LDS.
//     8 waves (4x2), wave tile 32x64 -> 2x4 v_wmma_f32_16x16x32_bf16 per
//     K-step (two K-steps per staged tile = 16 WMMA between barriers).
//     EPI=0: out = (acc + bias[c]) * aux[r]          (aux = rowScale[M])
//     EPI=1: out = acc + bias[c] + aux[r*N+c]        (aux = residual[M*N])
// ---------------------------------------------------------------------------
#define BKT    64
#define PITCH  72    // bf16 elements per LDS row (64 data + 8 pad)

template <int EPI>
__global__ __launch_bounds__(256) void gemm_bf16_kernel(const __bf16* __restrict__ A,
                                                        const __bf16* __restrict__ Bm,
                                                        const float*  __restrict__ bias,
                                                        const float*  __restrict__ aux,
                                                        float* __restrict__ out,
                                                        int M, int N, int K) {
    const int tid  = threadIdx.x;
    const int wave = tid >> 5;
    const int lane = tid & 31;
    const int wm   = wave & 3;   // m-offset wm*32
    const int wn   = wave >> 2;  // n-offset wn*64
    const int blockM = blockIdx.y * 128;
    const int blockN = blockIdx.x * 128;

    __shared__ __bf16 lA[2][128 * PITCH];
    __shared__ __bf16 lB[2][128 * PITCH];

    v8f acc[2][4];
#pragma unroll
    for (int mi = 0; mi < 2; ++mi)
#pragma unroll
        for (int ni = 0; ni < 4; ++ni)
            acc[mi][ni] = (v8f)0.f;

    const int mrow  = lane & 15;
    const int khalf = lane >> 4;

    v4u breg[4];
#if !USE_ASYNC_LDS
    v4u areg[4];
#endif

    // ---- staging helpers (each thread moves 4 x 16B per matrix per tile) ----
    auto stageA_issue = [&](int k0, int buf) {
#pragma unroll
        for (int it = 0; it < 4; ++it) {
            int e = it * 2048 + tid * 8;
            int row = e >> 6, kk = e & 63;
            const __bf16* g = A + (size_t)(blockM + row) * K + k0 + kk;
#if USE_ASYNC_LDS
            async_copy16(g, &lA[buf][row * PITCH + kk]);
#else
            areg[it] = *(const v4u*)g;
#endif
        }
        (void)buf;
    };
    auto stageA_commit = [&](int buf) {
#if !USE_ASYNC_LDS
#pragma unroll
        for (int it = 0; it < 4; ++it) {
            int e = it * 2048 + tid * 8;
            int row = e >> 6, kk = e & 63;
            *(v4u*)&lA[buf][row * PITCH + kk] = areg[it];
        }
#endif
        (void)buf;
    };
    auto loadB = [&](int k0) {
#pragma unroll
        for (int it = 0; it < 4; ++it) {
            int e = it * 2048 + tid * 8;
            int kk = e >> 7, n = e & 127;
            breg[it] = *(const v4u*)(Bm + (size_t)(k0 + kk) * N + blockN + n);
        }
    };
    auto storeB = [&](int buf) {   // transpose-on-store: LDS holds [n][k]
#pragma unroll
        for (int it = 0; it < 4; ++it) {
            int e = it * 2048 + tid * 8;
            int kk = e >> 7, n = e & 127;
            union { v4u q; __bf16 h[8]; } u; u.q = breg[it];
#pragma unroll
            for (int j = 0; j < 8; ++j) lB[buf][(n + j) * PITCH + kk] = u.h[j];
        }
    };
    auto compute = [&](int buf) {
#pragma unroll
        for (int s = 0; s < 2; ++s) {          // two 32-wide K-steps
            v16bf afrag[2], bfrag[4];
#pragma unroll
            for (int mi = 0; mi < 2; ++mi) {
                int rb = (wm * 32 + mi * 16 + mrow) * PITCH + s * 32;
                union { v4u q[2]; v16bf v; } fa;
                // A 16x32: lanes 0-15 K 0-7 & 16-23; lanes 16-31 K 8-15 & 24-31
                fa.q[0] = *(const v4u*)&lA[buf][rb + khalf * 8];
                fa.q[1] = *(const v4u*)&lA[buf][rb + 16 + khalf * 8];
                afrag[mi] = fa.v;
            }
#pragma unroll
            for (int ni = 0; ni < 4; ++ni) {
                int nb = (wn * 64 + ni * 16 + mrow) * PITCH + s * 32;
                union { v4u q[2]; v16bf v; } fb;
                // B 32x16: lanes 0-15 K 0-7 & 16-23; lanes 16-31 K 8-15 & 24-31
                fb.q[0] = *(const v4u*)&lB[buf][nb + khalf * 8];
                fb.q[1] = *(const v4u*)&lB[buf][nb + 16 + khalf * 8];
                bfrag[ni] = fb.v;
            }
#pragma unroll
            for (int mi = 0; mi < 2; ++mi)
#pragma unroll
                for (int ni = 0; ni < 4; ++ni)
                    acc[mi][ni] = __builtin_amdgcn_wmma_f32_16x16x32_bf16(
                        false, afrag[mi], false, bfrag[ni],
                        (short)0, acc[mi][ni], false, false);
        }
    };

    // ---- software-pipelined main loop (ping-pong LDS) ----
    const int nIter = K / BKT;
    stageA_issue(0, 0);
    loadB(0);
    stageA_commit(0);
    storeB(0);
#if USE_ASYNC_LDS
    __builtin_amdgcn_s_wait_asynccnt(0);
#endif
    __syncthreads();

    for (int i = 0; i < nIter; ++i) {
        const int cur = i & 1, nxt = cur ^ 1;
        const bool hasNext = (i + 1 < nIter);
        if (hasNext) {
            stageA_issue((i + 1) * BKT, nxt);   // async DMA / global->reg prefetch
            loadB((i + 1) * BKT);
        }
        compute(cur);                            // 16 WMMAs hide the prefetch
        if (hasNext) {
            stageA_commit(nxt);
            storeB(nxt);
        }
#if USE_ASYNC_LDS
        __builtin_amdgcn_s_wait_asynccnt(0);
#endif
        __syncthreads();
    }

    // ---- branch-free epilogue (EPI is a compile-time constant) ----
    const int col16 = lane & 15;
    const int rsel  = lane >> 4;
#pragma unroll
    for (int mi = 0; mi < 2; ++mi)
#pragma unroll
        for (int ni = 0; ni < 4; ++ni) {
            union { v8f v; float f[8]; } u; u.v = acc[mi][ni];
            const int cgl  = blockN + wn * 64 + ni * 16 + col16;
            const int row0 = blockM + wm * 32 + mi * 16 + rsel * 8;
            const float bc = bias[cgl];
            size_t idx = (size_t)row0 * N + cgl;
            if (EPI == 0) {
#pragma unroll
                for (int i = 0; i < 8; ++i) {
                    out[idx] = (u.f[i] + bc) * aux[row0 + i];
                    idx += (size_t)N;
                }
            } else {
#pragma unroll
                for (int i = 0; i < 8; ++i) {
                    out[idx] = u.f[i] + bc + aux[idx];
                    idx += (size_t)N;
                }
            }
        }
}

// ---------------------------------------------------------------------------
// K3: local inclusive scan over each L-chunk (in place) + chunk totals
// ---------------------------------------------------------------------------
__global__ __launch_bounds__(256) void scan_local_kernel(float* __restrict__ bound,
                                                         float* __restrict__ chunkSum) {
    const int d = blockIdx.x * 256 + threadIdx.x;
    const int c = blockIdx.y;
    const int b = blockIdx.z;
    size_t base = ((size_t)b * LL + (size_t)c * CHUNK) * DIMD + d;
    float run = 0.f;
    for (int i = 0; i < CHUNK; ++i) {
        run += bound[base + (size_t)i * DIMD];
        bound[base + (size_t)i * DIMD] = run;
    }
    chunkSum[((size_t)b * NCHUNK + c) * DIMD + d] = run;
}

// K4: exclusive scan of chunk totals over c (in place)
__global__ __launch_bounds__(256) void scan_chunks_kernel(float* __restrict__ chunkSum) {
    const int d = blockIdx.x * 256 + threadIdx.x;
    const int b = blockIdx.y;
    size_t base = (size_t)b * NCHUNK * DIMD + d;
    float run = 0.f;
    for (int c = 0; c < NCHUNK; ++c) {
        float t = chunkSum[base + (size_t)c * DIMD];
        chunkSum[base + (size_t)c * DIMD] = run;
        run += t;
    }
}

// ---------------------------------------------------------------------------
// K5: fused add-chunk-offset, /sqrt(L), layernorm, emit h in bf16
// ---------------------------------------------------------------------------
__global__ __launch_bounds__(256) void ln_kernel(const float* __restrict__ bound,
                                                 const float* __restrict__ chunkOff,
                                                 const float* __restrict__ g,
                                                 const float* __restrict__ bln,
                                                 __bf16* __restrict__ h) {
    const int m = blockIdx.x;
    const int b = m >> 12;          // / LL
    const int l = m & (LL - 1);
    const int c = l >> 6;           // / CHUNK
    const float* row = bound + (size_t)m * DIMD;
    const float* off = chunkOff + ((size_t)b * NCHUNK + c) * DIMD;
    const float inv = 0.015625f;    // 1/sqrt(4096)

    float v[8];
    float sum = 0.f, sq = 0.f;
#pragma unroll
    for (int i = 0; i < 8; ++i) {
        int j = threadIdx.x + i * 256;
        v[i] = (row[j] + off[j]) * inv;
        sum += v[i];
        sq  += v[i] * v[i];
    }
    for (int o = 16; o; o >>= 1) { sum += __shfl_xor(sum, o); sq += __shfl_xor(sq, o); }

    __shared__ float rs[8], rq[8];
    const int wave = threadIdx.x >> 5, lane = threadIdx.x & 31;
    if (lane == 0) { rs[wave] = sum; rq[wave] = sq; }
    __syncthreads();
    if (wave == 0) {
        float s2 = (lane < 8) ? rs[lane] : 0.f;
        float q2 = (lane < 8) ? rq[lane] : 0.f;
        for (int o = 4; o; o >>= 1) { s2 += __shfl_xor(s2, o); q2 += __shfl_xor(q2, o); }
        if (lane == 0) { rs[0] = s2; rq[0] = q2; }
    }
    __syncthreads();
    const float mu  = rs[0] * (1.f / DIMD);
    const float var = rq[0] * (1.f / DIMD) - mu * mu;
    const float rst = rsqrtf(var + 1e-5f);
#pragma unroll
    for (int i = 0; i < 8; ++i) {
        int j = threadIdx.x + i * 256;
        h[(size_t)m * DIMD + j] = (__bf16)((v[i] - mu) * rst * g[j] + bln[j]);
    }
}

// ---------------------------------------------------------------------------
// Host-side launcher
// ---------------------------------------------------------------------------
extern "C" void kernel_launch(void* const* d_in, const int* in_sizes, int n_in,
                              void* d_out, int out_size, void* d_ws, size_t ws_size,
                              hipStream_t stream) {
    const float* x   = (const float*)d_in[0];
    const float* Wp  = (const float*)d_in[1];
    const float* bp  = (const float*)d_in[2];
    const float* Wa  = (const float*)d_in[3];
    const float* ba  = (const float*)d_in[4];
    const float* Wv  = (const float*)d_in[5];
    const float* bv  = (const float*)d_in[6];
    const float* lng = (const float*)d_in[7];
    const float* lnb = (const float*)d_in[8];
    const float* Wo  = (const float*)d_in[9];
    const float* bo  = (const float*)d_in[10];
    float* out = (float*)d_out;

    // workspace layout (bytes); h_bf16 aliases x_bf16 (dead after value GEMM)
    char* ws = (char*)d_ws;
    const size_t X16   = 0;                                   // 64 MB
    const size_t WV16  = X16  + (size_t)MTOT * DIMD * 2;      // 8 MB
    const size_t WO16  = WV16 + (size_t)DIMD * DIMD * 2;      // 8 MB
    const size_t SBUF  = WO16 + (size_t)DIMD * DIMD * 2;      // 64 KB
    const size_t BOUND = SBUF + (size_t)MTOT * 4;             // 128 MB
    const size_t CSUM  = BOUND + (size_t)MTOT * DIMD * 4;     // 2 MB
    __bf16* x16  = (__bf16*)(ws + X16);
    __bf16* wv16 = (__bf16*)(ws + WV16);
    __bf16* wo16 = (__bf16*)(ws + WO16);
    float*  sbuf = (float*)(ws + SBUF);
    float*  bnd  = (float*)(ws + BOUND);
    float*  csum = (float*)(ws + CSUM);
    __bf16* h16  = (__bf16*)(ws + X16);   // alias

    // K0: conversions
    {
        long n4x = (long)MTOT * DIMD / 4;
        cvt_bf16_kernel<<<dim3((unsigned)(n4x / 256)), 256, 0, stream>>>(x, x16, n4x);
        long n4w = (long)DIMD * DIMD / 4;
        cvt_bf16_kernel<<<dim3((unsigned)(n4w / 256)), 256, 0, stream>>>(Wv, wv16, n4w);
        cvt_bf16_kernel<<<dim3((unsigned)(n4w / 256)), 256, 0, stream>>>(Wo, wo16, n4w);
    }
    // K1: per-token phasor scalar
    phase_amp_kernel<<<dim3(MTOT / 8), 256, 0, stream>>>(x, Wp, bp, Wa, ba, sbuf);

    // K2: bound = (x @ Wv + bv) * s[row]
    gemm_bf16_kernel<0><<<dim3(DIMD / 128, MTOT / 128), 256, 0, stream>>>(
        x16, wv16, bv, sbuf, bnd, MTOT, DIMD, DIMD);

    // K3/K4: hierarchical cumsum over L
    scan_local_kernel<<<dim3(DIMD / 256, NCHUNK, BB), 256, 0, stream>>>(bnd, csum);
    scan_chunks_kernel<<<dim3(DIMD / 256, BB), 256, 0, stream>>>(csum);

    // K5: layernorm -> h (bf16)
    ln_kernel<<<dim3(MTOT), 256, 0, stream>>>(bnd, csum, lng, lnb, h16);

    // K6: out = x + h @ Wo + bo
    gemm_bf16_kernel<1><<<dim3(DIMD / 128, MTOT / 128), 256, 0, stream>>>(
        h16, wo16, bo, x, out, MTOT, DIMD, DIMD);

    (void)in_sizes; (void)n_in; (void)out_size; (void)ws_size;
}
